// APSGNNModel_19610820674284
// MI455X (gfx1250) — matrix operand order repaired and checked by reading, not compiled, not run on hardware
//
#include <hip/hip_runtime.h>

#define Bx   512
#define Wx   4
#define KDx  64
#define NCx  32
#define Dx   256
#define NNx  16
#define NHx  8
#define ADx  32
#define HOPSx 4
#define DHx  32
#define DFFx 1024
#define Px   (Bx*Wx + Bx)        /* 2560 packets */
#define MAXT (Px/16 + NNx)       /* 176 worst-case row tiles */
#define APAD 8                   /* LDS row pad (halves) -> bank-conflict free */
#define VROW 40                  /* transposed-V LDS row stride (32 + 8 pad)   */

typedef __attribute__((ext_vector_type(16))) _Float16 v16h;
typedef __attribute__((ext_vector_type(8)))  float    v8f;

union AFrag  { v16h v; uint4 q[2]; };
union BFrag  { v16h v; unsigned u[8]; };
union HPack8 { _Float16 h[8]; unsigned u[4]; uint4 q; };

__device__ __forceinline__ v8f wmma_f16(v16h a, v16h b, v8f c) {
  // D = A(16x32 f16) * B(32x16 f16) + C(16x16 f32)
  return __builtin_amdgcn_wmma_f32_16x16x32_f16(false, a, false, b, (short)0, c, false, false);
}

__device__ __forceinline__ float blockSum256(float v, float* red) {
  int t = threadIdx.x;
  red[t] = v; __syncthreads();
  #pragma unroll
  for (int s = 128; s > 0; s >>= 1) { if (t < s) red[t] += red[t + s]; __syncthreads(); }
  float r = red[0]; __syncthreads();
  return r;
}

// ---------------------------------------------------------------------------
// Pack f32 weights [NN][K][N] into WMMA-B f16 tiles:
//   tile = (node, kt, nt), 512 halves/tile, lane reads 16 contiguous halves.
//   B layout (32x16): lanes 0-15 hold K=0..15 (half e -> K=e), lanes 16-31 K=16..31.
__global__ void pack_b_kernel(const float* __restrict__ src, _Float16* __restrict__ dst,
                              int Ktot, int N) {
  long g = (long)blockIdx.x * 256 + threadIdx.x;
  long total = (long)NNx * Ktot * N;
  if (g >= total) return;
  int tile = (int)(g >> 9), rem = (int)(g & 511);
  int lane = rem >> 4, e = rem & 15;
  int ktiles = Ktot / 32, ntiles = N / 16;
  int nodei = tile / (ktiles * ntiles);
  int rem2  = tile % (ktiles * ntiles);
  int kt = rem2 / ntiles, nt = rem2 % ntiles;
  int hi = lane >> 4;
  int K  = kt * 32 + e + (hi ? 16 : 0);
  int n  = nt * 16 + (lane & 15);
  dst[g] = (_Float16)src[(long)nodei * Ktot * N + (long)K * N + n];
}

// ---------------------------------------------------------------------------
__global__ void encode_kernel(const float* __restrict__ qk, const float* __restrict__ wk,
                              const float* __restrict__ kpw, const float* __restrict__ kpb,
                              const float* __restrict__ ce, const float* __restrict__ re,
                              const float* __restrict__ sne, const float* __restrict__ ilg,
                              const float* __restrict__ ilb, const int* __restrict__ qsn,
                              const int* __restrict__ wl, const int* __restrict__ wsn,
                              float* __restrict__ h, int* __restrict__ node) {
  __shared__ float sk[KDx];
  __shared__ float red[256];
  int p = blockIdx.x, t = threadIdx.x;
  bool isW = p < Bx * Wx;
  const float* key; int sn, label = 0;
  if (isW) { key = wk + (long)p * KDx; label = wl[p]; sn = wsn[p]; }
  else     { int qi = p - Bx * Wx; key = qk + (long)qi * KDx; sn = qsn[qi]; }
  if (t < KDx) sk[t] = key[t];
  __syncthreads();
  float acc = kpb[t];
  #pragma unroll 8
  for (int k = 0; k < KDx; ++k) acc += sk[k] * kpw[k * Dx + t];
  acc += sne[sn * Dx + t];
  acc += isW ? (ce[label * Dx + t] + re[0 * Dx + t]) : re[1 * Dx + t];
  float mean = blockSum256(acc, red) * (1.f / Dx);
  float d    = acc - mean;
  float var  = blockSum256(d * d, red) * (1.f / Dx);
  float ln   = d * rsqrtf(var + 1e-5f);
  float sw = 0.f;
  if (t < KDx) sw = sk[t];
  else if (isW && t < KDx + NCx) sw = (t - KDx == label) ? 1.f : 0.f;
  h[(long)p * Dx + t] = ln * ilg[t] + ilb[t] + sw;
  if (t == 0) node[p] = sn;
}

// ---------------------------------------------------------------------------
// Deterministic stable counting sort by node; builds segment offsets + tile map.
__global__ void sort_kernel(const int* __restrict__ node, int* __restrict__ pos,
                            int* __restrict__ order, int* __restrict__ segOffG,
                            int* __restrict__ tileN, int* __restrict__ tileR,
                            int* __restrict__ numT) {
  __shared__ int snd[Px];
  __shared__ int cnt[NNx];
  __shared__ int off[NNx + 1];
  int t = threadIdx.x;
  for (int i = t; i < Px; i += 256) snd[i] = node[i];
  if (t < NNx) cnt[t] = 0;
  __syncthreads();
  for (int i = t; i < Px; i += 256) atomicAdd(&cnt[snd[i]], 1);
  __syncthreads();
  if (t == 0) { int s = 0; for (int n = 0; n < NNx; ++n) { off[n] = s; s += cnt[n]; } off[NNx] = s; }
  __syncthreads();
  for (int p = t; p < Px; p += 256) {
    int nn = snd[p], r = 0;
    for (int q = 0; q < p; ++q) r += (snd[q] == nn) ? 1 : 0;
    int ps = off[nn] + r;
    pos[p] = ps; order[ps] = p;
  }
  if (t == 0) {
    for (int n = 0; n <= NNx; ++n) segOffG[n] = off[n];
    int tc = 0;
    for (int n = 0; n < NNx; ++n)
      for (int r0 = off[n]; r0 < off[n + 1]; r0 += 16) { tileN[tc] = n; tileR[tc] = r0; ++tc; }
    numT[0] = tc;
  }
}

// ---------------------------------------------------------------------------
__global__ void ln_kernel(const float* __restrict__ h, const float* __restrict__ g,
                          const float* __restrict__ b, const int* __restrict__ node,
                          const int* __restrict__ pos, _Float16* __restrict__ xs) {
  __shared__ float red[256];
  int p = blockIdx.x, t = threadIdx.x;
  float hv   = h[(long)p * Dx + t];
  float mean = blockSum256(hv, red) * (1.f / Dx);
  float d    = hv - mean;
  float var  = blockSum256(d * d, red) * (1.f / Dx);
  int n = node[p];
  float val = d * rsqrtf(var + 1e-5f) * g[n * Dx + t] + b[n * Dx + t];
  xs[(long)pos[p] * Dx + t] = (_Float16)val;
}

// ---------------------------------------------------------------------------
// Node-segment GEMM: rows = sorted packets of one node, B = packed per-node weights.
// Block = 8 waves; each wave owns TWO adjacent 16x16 output tiles sharing one A
// fragment (2 wmma per A load, conflict-free padded LDS). Modes:
//   0: Y(f16) = X*W + bias          (qkv)
//   1: Y(f16) = gelu(X*W + bias)    (fc1)
//   2: h[order[r]] += X*W + bias    (wo, fc2)
__global__ void gemm_node_kernel(const _Float16* __restrict__ X, const _Float16* __restrict__ Bpk,
                                 const float* __restrict__ bias, _Float16* __restrict__ Yh,
                                 float* __restrict__ hacc, const int* __restrict__ order,
                                 const int* __restrict__ segOff, const int* __restrict__ tileN,
                                 const int* __restrict__ tileR, const int* __restrict__ numT,
                                 int K, int N, int mode) {
  __shared__ _Float16 lsA[16 * (DFFx + APAD)];
  if ((int)blockIdx.x >= numT[0]) return;
  int nodei  = tileN[blockIdx.x];
  int r0     = tileR[blockIdx.x];
  int segEnd = segOff[nodei + 1];
  int stride = K + APAD;

  int vecs = 16 * K / 8;
  for (int i = threadIdx.x; i < vecs; i += 256) {
    int row = (i * 8) / K, offh = (i * 8) % K;
    int r = r0 + row;
    uint4 v = {0u, 0u, 0u, 0u};
    if (r < segEnd) v = *(const uint4*)(X + (long)r * K + offh);
    *(uint4*)(lsA + row * stride + offh) = v;
  }
  __syncthreads();

  int lane = threadIdx.x & 31, wave = threadIdx.x >> 5, hi = lane >> 4;
  int ntg0 = (blockIdx.y * 8 + wave) * 2;          // two adjacent column tiles
  int ktiles = K / 32, ntiles = N / 16;
  v8f acc0 = {0.f, 0.f, 0.f, 0.f, 0.f, 0.f, 0.f, 0.f};
  v8f acc1 = {0.f, 0.f, 0.f, 0.f, 0.f, 0.f, 0.f, 0.f};
  long btileBase = (long)nodei * ktiles * ntiles + ntg0;
  for (int kt = 0; kt < ktiles; ++kt) {
    AFrag a;
    int b0 = (lane & 15) * stride + kt * 32 + (hi ? 16 : 0);
    a.q[0] = *(const uint4*)(lsA + b0);
    a.q[1] = *(const uint4*)(lsA + b0 + 8);
    const _Float16* bp = Bpk + (btileBase + (long)kt * ntiles) * 512 + lane * 16;
    const v16h bf0 = *(const v16h*)(bp);
    const v16h bf1 = *(const v16h*)(bp + 512);
    if (kt + 1 < ktiles) {
      const _Float16* bn = Bpk + (btileBase + (long)(kt + 1) * ntiles) * 512 + lane * 16;
      __builtin_prefetch(bn, 0, 1);
      __builtin_prefetch(bn + 512, 0, 1);
    }
    acc0 = wmma_f16(a.v, bf0, acc0);
    acc1 = wmma_f16(a.v, bf1, acc1);
  }

  int col0 = ntg0 * 16 + (lane & 15);
  float bv0 = bias[nodei * N + col0];
  float bv1 = bias[nodei * N + col0 + 16];
  #pragma unroll
  for (int e = 0; e < 8; ++e) {
    int r = r0 + e + (hi ? 8 : 0);
    if (r >= segEnd) continue;
    float v0 = acc0[e] + bv0;
    float v1 = acc1[e] + bv1;
    if (mode == 0) {
      Yh[(long)r * N + col0]      = (_Float16)v0;
      Yh[(long)r * N + col0 + 16] = (_Float16)v1;
    } else if (mode == 1) {
      float g0 = 0.5f * v0 * (1.f + tanhf(0.7978845608f * (v0 + 0.044715f * v0 * v0 * v0)));
      float g1 = 0.5f * v1 * (1.f + tanhf(0.7978845608f * (v1 + 0.044715f * v1 * v1 * v1)));
      Yh[(long)r * N + col0]      = (_Float16)g0;
      Yh[(long)r * N + col0 + 16] = (_Float16)g1;
    } else {
      long pk = (long)order[r] * Dx;
      hacc[pk + col0]      += v0;
      hacc[pk + col0 + 16] += v1;
    }
  }
}

// ---------------------------------------------------------------------------
// Block-diagonal flash attention. One block per 16-query tile; wave = head.
// S^T = K_tile(16x32) * Q^T(32x16): per-lane column = one query -> lane-local softmax
// (stats exchanged with the lane^16 partner). P*V via two WMMAs per 32-key chunk
// using LDS-transposed V. All LDS rows padded for conflict-free b128 reads.
__global__ void attn_kernel(const _Float16* __restrict__ qkv, _Float16* __restrict__ ao,
                            const int* __restrict__ segOff, const int* __restrict__ tileN,
                            const int* __restrict__ tileR, const int* __restrict__ numT) {
  __shared__ _Float16 lsK[32 * (Dx + APAD)];  // 32 keys x 256 (all heads), padded
  __shared__ _Float16 lsV[Dx * VROW];         // V transposed: [dh][key], padded
  if ((int)blockIdx.x >= numT[0]) return;
  int nodei = tileN[blockIdx.x];
  int r0    = tileR[blockIdx.x];
  int s0 = segOff[nodei], segEnd = segOff[nodei + 1];

  int lane = threadIdx.x & 31, hd = threadIdx.x >> 5, hi = lane >> 4, qi = lane & 15;
  int qrow = r0 + qi;
  // Q as WMMA-B operand: lane<16 -> dh 0..15, lane>=16 -> dh 16..31 (contiguous halves)
  const v16h qb = *(const v16h*)(qkv + (long)qrow * 768 + hd * DHx + (hi ? 16 : 0));

  float m = -1e30f, den = 0.f;
  v8f oA = {0.f, 0.f, 0.f, 0.f, 0.f, 0.f, 0.f, 0.f};
  v8f oB = {0.f, 0.f, 0.f, 0.f, 0.f, 0.f, 0.f, 0.f};

  for (int kc = s0; kc < segEnd; kc += 32) {
    __syncthreads();
    for (int i = threadIdx.x; i < 32 * Dx / 8; i += 256) {   // K rows (k slice)
      int row = i / (Dx / 8), offh = (i % (Dx / 8)) * 8;
      int r = kc + row;
      uint4 v = {0u, 0u, 0u, 0u};
      if (r < segEnd) v = *(const uint4*)(qkv + (long)r * 768 + Dx + offh);
      *(uint4*)(lsK + row * (Dx + APAD) + offh) = v;
    }
    for (int i = threadIdx.x; i < 32 * Dx / 8; i += 256) {   // V transposed
      int key = i / (Dx / 8), dh0 = (i % (Dx / 8)) * 8;
      int r = kc + key;
      if (r < segEnd) {
        const _Float16* src = qkv + (long)r * 768 + 2 * Dx + dh0;
        #pragma unroll
        for (int j = 0; j < 8; ++j) lsV[(dh0 + j) * VROW + key] = src[j];
      } else {
        #pragma unroll
        for (int j = 0; j < 8; ++j) lsV[(dh0 + j) * VROW + key] = (_Float16)0.f;
      }
    }
    __syncthreads();

    v8f sc[2];
    #pragma unroll
    for (int sub = 0; sub < 2; ++sub) {
      AFrag ka;
      int b0 = (sub * 16 + qi) * (Dx + APAD) + hd * DHx + (hi ? 16 : 0);
      ka.q[0] = *(const uint4*)(lsK + b0);
      ka.q[1] = *(const uint4*)(lsK + b0 + 8);
      v8f z = {0.f, 0.f, 0.f, 0.f, 0.f, 0.f, 0.f, 0.f};
      sc[sub] = wmma_f16(ka.v, qb, z);
    }

    float sv[16]; float tm = -1e30f;
    #pragma unroll
    for (int idx = 0; idx < 16; ++idx) {
      int sub = idx >> 3, e = idx & 7;
      float x = sc[sub][e] * 0.17677669529f;               // 1/sqrt(32)
      int krow = kc + sub * 16 + (hi ? 8 : 0) + e;
      if (krow >= segEnd) x = -1e30f;
      sv[idx] = x; tm = fmaxf(tm, x);
    }
    tm = fmaxf(tm, __shfl_xor(tm, 16));
    float newm = fmaxf(m, tm);
    float corr = __expf(m - newm);
    #pragma unroll
    for (int e = 0; e < 8; ++e) { oA[e] *= corr; oB[e] *= corr; }
    den *= corr;

    HPack8 P0, P1; float ps = 0.f;
    #pragma unroll
    for (int idx = 0; idx < 16; ++idx) {
      float pe = __expf(sv[idx] - newm);
      ps += pe;
      if (idx < 8) P0.h[idx] = (_Float16)pe; else P1.h[idx - 8] = (_Float16)pe;
    }
    ps += __shfl_xor(ps, 16);
    den += ps; m = newm;

    // Assemble probs as B operand (32 keys x 16 queries) via lane-pair exchange.
    BFrag pb;
    #pragma unroll
    for (int j = 0; j < 4; ++j) {
      unsigned o0 = __shfl_xor(P0.u[j], 16);
      unsigned o1 = __shfl_xor(P1.u[j], 16);
      if (hi == 0) { pb.u[j] = P0.u[j]; pb.u[4 + j] = o0; }
      else         { pb.u[j] = o1;      pb.u[4 + j] = P1.u[j]; }
    }

    AFrag va, vb;
    int d0 = (hd * DHx + qi) * VROW + (hi ? 16 : 0);        // dh rows 0..15 of this head
    va.q[0] = *(const uint4*)(lsV + d0);
    va.q[1] = *(const uint4*)(lsV + d0 + 8);
    int d1 = (hd * DHx + 16 + qi) * VROW + (hi ? 16 : 0);   // dh rows 16..31
    vb.q[0] = *(const uint4*)(lsV + d1);
    vb.q[1] = *(const uint4*)(lsV + d1 + 8);
    oA = wmma_f16(va.v, pb.v, oA);
    oB = wmma_f16(vb.v, pb.v, oB);
  }

  if (qrow < segEnd) {
    float inv = 1.f / den;
    HPack8 o1, o2;
    #pragma unroll
    for (int e = 0; e < 8; ++e) { o1.h[e] = (_Float16)(oA[e] * inv); o2.h[e] = (_Float16)(oB[e] * inv); }
    long baseo = (long)qrow * Dx + hd * DHx + (hi ? 8 : 0);
    *(uint4*)(ao + baseo)      = o1.q;
    *(uint4*)(ao + baseo + 16) = o2.q;
  }
}

// ---------------------------------------------------------------------------
__global__ void route_kernel(float* __restrict__ h, int* __restrict__ node,
                             const float* __restrict__ dw, const float* __restrict__ db,
                             const float* __restrict__ dirw, const float* __restrict__ dirb,
                             const float* __restrict__ magw, const float* __restrict__ magb,
                             const float* __restrict__ at) {
  __shared__ float hl[Dx];
  __shared__ float red[256];
  __shared__ float dirS[ADx];
  __shared__ float scS[NNx];
  int p = blockIdx.x, t = threadIdx.x;
  hl[t] = h[(long)p * Dx + t];
  __syncthreads();
  float dv = db[t];
  for (int k = 0; k < Dx; ++k) dv += hl[k] * dw[k * Dx + t];
  float mg = blockSum256(hl[t] * magw[t], red) + magb[0];
  if (t < ADx) { float x = dirb[t]; for (int k = 0; k < Dx; ++k) x += hl[k] * dirw[k * Dx + t]; dirS[t] = x; }
  __syncthreads();
  if (t < NNx) { float s = 0.f; for (int a = 0; a < ADx; ++a) s += dirS[a] * at[t * ADx + a]; scS[t] = s; }
  __syncthreads();
  if (t == 0) {
    float b = scS[0]; int bi = 0;
    for (int n = 1; n < NNx; ++n) if (scS[n] > b) { b = scS[n]; bi = n; }
    node[p] = bi;
  }
  float gate = 1.f / (1.f + __expf(-mg));
  h[(long)p * Dx + t] = hl[t] + dv * gate;
}

// ---------------------------------------------------------------------------
__global__ void out_kernel(const float* __restrict__ h, const float* __restrict__ ow,
                           const float* __restrict__ ob, float* __restrict__ out) {
  __shared__ float hl[Dx];
  int p = blockIdx.x, t = threadIdx.x;
  hl[t] = h[(long)p * Dx + t];
  __syncthreads();
  if (t < NCx) {
    float o = ob[t];
    for (int k = 0; k < Dx; ++k) o += hl[k] * ow[k * NCx + t];
    out[(long)p * NCx + t] = o;
  }
}

// ---------------------------------------------------------------------------
extern "C" void kernel_launch(void* const* d_in, const int* in_sizes, int n_in,
                              void* d_out, int out_size, void* d_ws, size_t ws_size,
                              hipStream_t stream) {
  const float* query_keys       = (const float*)d_in[0];
  const float* writer_keys      = (const float*)d_in[1];
  const float* key_proj_w       = (const float*)d_in[2];
  const float* key_proj_b       = (const float*)d_in[3];
  const float* class_embed      = (const float*)d_in[4];
  const float* role_embed       = (const float*)d_in[5];
  const float* start_node_embed = (const float*)d_in[6];
  const float* input_ln_g       = (const float*)d_in[7];
  const float* input_ln_b       = (const float*)d_in[8];
  const float* ln1_g            = (const float*)d_in[9];
  const float* ln1_b            = (const float*)d_in[10];
  const float* wqkv             = (const float*)d_in[11];
  const float* bqkv             = (const float*)d_in[12];
  const float* wo               = (const float*)d_in[13];
  const float* bo               = (const float*)d_in[14];
  const float* ln2_g            = (const float*)d_in[15];
  const float* ln2_b            = (const float*)d_in[16];
  const float* w_fc1            = (const float*)d_in[17];
  const float* b_fc1            = (const float*)d_in[18];
  const float* w_fc2            = (const float*)d_in[19];
  const float* b_fc2            = (const float*)d_in[20];
  const float* delta_w          = (const float*)d_in[21];
  const float* delta_b          = (const float*)d_in[22];
  const float* dir_w            = (const float*)d_in[23];
  const float* dir_b            = (const float*)d_in[24];
  const float* mag_w            = (const float*)d_in[25];
  const float* mag_b            = (const float*)d_in[26];
  const float* out_w            = (const float*)d_in[27];
  const float* out_b            = (const float*)d_in[28];
  const float* address_table    = (const float*)d_in[29];
  const int*   qsn              = (const int*)d_in[30];
  const int*   wl               = (const int*)d_in[31];
  const int*   wsn              = (const int*)d_in[32];
  (void)in_sizes; (void)n_in; (void)out_size; (void)ws_size;

  char* base = (char*)d_ws;
  size_t cur = 0;
  auto alloc = [&](size_t bytes) -> void* {
    void* p = base + cur; cur = (cur + bytes + 255) & ~(size_t)255; return p;
  };
  float*     h     = (float*)alloc((size_t)Px * Dx * 4);
  _Float16*  xs    = (_Float16*)alloc((size_t)(Px + 16) * Dx * 2);
  _Float16*  qkvb  = (_Float16*)alloc((size_t)(Px + 16) * 768 * 2);
  _Float16*  aob   = (_Float16*)alloc((size_t)(Px + 16) * Dx * 2);
  _Float16*  gbuf  = (_Float16*)alloc((size_t)(Px + 16) * DFFx * 2);
  int*       node  = (int*)alloc(Px * 4);
  int*       pos   = (int*)alloc(Px * 4);
  int*       order = (int*)alloc(Px * 4);
  int*       segOff = (int*)alloc((NNx + 1) * 4);
  int*       tileN  = (int*)alloc(MAXT * 4);
  int*       tileR  = (int*)alloc(MAXT * 4);
  int*       numT   = (int*)alloc(4);
  _Float16*  pQKV  = (_Float16*)alloc((size_t)NNx * Dx * 768 * 2);
  _Float16*  pWO   = (_Float16*)alloc((size_t)NNx * Dx * Dx * 2);
  _Float16*  pF1   = (_Float16*)alloc((size_t)NNx * Dx * DFFx * 2);
  _Float16*  pF2   = (_Float16*)alloc((size_t)NNx * DFFx * Dx * 2);

  // Pre-pack weights into WMMA-B f16 tiles (resident in L2 for all 4 hops).
  pack_b_kernel<<<(NNx * Dx * 768) / 256, 256, 0, stream>>>(wqkv, pQKV, Dx, 3 * Dx);
  pack_b_kernel<<<(NNx * Dx * Dx) / 256, 256, 0, stream>>>(wo, pWO, Dx, Dx);
  pack_b_kernel<<<(NNx * Dx * DFFx) / 256, 256, 0, stream>>>(w_fc1, pF1, Dx, DFFx);
  pack_b_kernel<<<(NNx * DFFx * Dx) / 256, 256, 0, stream>>>(w_fc2, pF2, DFFx, Dx);

  encode_kernel<<<Px, 256, 0, stream>>>(query_keys, writer_keys, key_proj_w, key_proj_b,
      class_embed, role_embed, start_node_embed, input_ln_g, input_ln_b, qsn, wl, wsn, h, node);

  for (int hop = 0; hop < HOPSx; ++hop) {
    sort_kernel<<<1, 256, 0, stream>>>(node, pos, order, segOff, tileN, tileR, numT);
    ln_kernel<<<Px, 256, 0, stream>>>(h, ln1_g, ln1_b, node, pos, xs);
    gemm_node_kernel<<<dim3(MAXT, 3), 256, 0, stream>>>(xs, pQKV, bqkv, qkvb, nullptr,
        order, segOff, tileN, tileR, numT, Dx, 3 * Dx, 0);
    attn_kernel<<<MAXT, 256, 0, stream>>>(qkvb, aob, segOff, tileN, tileR, numT);
    gemm_node_kernel<<<dim3(MAXT, 1), 256, 0, stream>>>(aob, pWO, bo, nullptr, h,
        order, segOff, tileN, tileR, numT, Dx, Dx, 2);
    ln_kernel<<<Px, 256, 0, stream>>>(h, ln2_g, ln2_b, node, pos, xs);
    gemm_node_kernel<<<dim3(MAXT, 4), 256, 0, stream>>>(xs, pF1, b_fc1, gbuf, nullptr,
        order, segOff, tileN, tileR, numT, Dx, DFFx, 1);
    gemm_node_kernel<<<dim3(MAXT, 1), 256, 0, stream>>>(gbuf, pF2, b_fc2, nullptr, h,
        order, segOff, tileN, tileR, numT, DFFx, Dx, 2);
    route_kernel<<<Px, 256, 0, stream>>>(h, node, delta_w, delta_b, dir_w, dir_b,
        mag_w, mag_b, address_table);
  }
  out_kernel<<<Px, 256, 0, stream>>>(h, out_w, out_b, (float*)d_out);
}
